// AttentionBlock_39822936769010
// MI455X (gfx1250) — matrix operand order repaired and checked by reading, not compiled, not run on hardware
//
#include <hip/hip_runtime.h>

// ---------------- problem constants ----------------
#define BATCH 8
#define CH    256          // channels (= attention "head dim")
#define NPOS  4096         // H*W
#define NGRP  32
#define CPG   (CH / NGRP)  // 8
#define SCALE 0.0625f      // CH^-0.5 = 1/16

typedef __bf16 bf16;
typedef __attribute__((ext_vector_type(16))) __bf16 v16bf;
typedef __attribute__((ext_vector_type(8)))  float  v8f;

struct U32x8 { uint4 lo; uint4 hi; };

__device__ __forceinline__ v8f vzero8() {
    v8f z = {0.f, 0.f, 0.f, 0.f, 0.f, 0.f, 0.f, 0.f};
    return z;
}

__device__ __forceinline__ bf16 f2bf(float f) {
    unsigned u = __float_as_uint(f);
    unsigned r = u + 0x7FFFu + ((u >> 16) & 1u);     // round-to-nearest-even
    unsigned short h = (unsigned short)(r >> 16);
    return __builtin_bit_cast(bf16, h);
}

// ---- WMMA fragment loaders (wave32, 16-bit layouts per CDNA5 ISA 7.12.2) ----
// A: 16x32 (MxK), src row-major [M][ld], lane L: m=L%16, half=L/16
//    v0..v3 <- K = half*8 + 0..7 ; v4..v7 <- K = 16 + half*8 + 0..7
__device__ __forceinline__ v16bf load_frag_a(const bf16* src, int ld) {
    int lane = threadIdx.x & 31;
    int m    = lane & 15;
    int half = lane >> 4;
    const bf16* p = src + (size_t)m * ld + half * 8;
    U32x8 t;
    t.lo = *(const uint4*)(p);
    t.hi = *(const uint4*)(p + 16);
    return __builtin_bit_cast(v16bf, t);
}

// B: 32x16 (KxN), src stored [N][ld] (per-column contiguous K), lane L: n=L%16
//    v0..v3 <- K = half*8 + 0..7 ; v4..v7 <- K = 16 + half*8 + 0..7
__device__ __forceinline__ v16bf load_frag_b(const bf16* src, long ld) {
    int lane = threadIdx.x & 31;
    int n    = lane & 15;
    int half = lane >> 4;
    const bf16* p = src + (size_t)n * ld + half * 8;
    U32x8 t;
    t.lo = *(const uint4*)(p);
    t.hi = *(const uint4*)(p + 16);
    return __builtin_bit_cast(v16bf, t);
}

__device__ __forceinline__ v8f wmma_bf16(v16bf a, v16bf b, v8f c) {
    return __builtin_amdgcn_wmma_f32_16x16x32_bf16(false, a, false, b,
                                                   (short)0, c, false, false);
}

// ---------------- K1: GroupNorm statistics ----------------
// one block per (b,g); each group's data is CPG*NPOS contiguous floats
__global__ void gn_stats_kernel(const float* __restrict__ x,
                                float* __restrict__ stats) {
    const int bg = blockIdx.x;                // 0 .. BATCH*NGRP-1
    const int n  = CPG * NPOS;                // 32768
    const float* p = x + (size_t)bg * n;
    float s = 0.f, ss = 0.f;
    for (int i = threadIdx.x; i < n; i += 256) {
        float v = p[i];
        s += v; ss += v * v;
    }
    __shared__ float sh0[256], sh1[256];
    sh0[threadIdx.x] = s; sh1[threadIdx.x] = ss;
    __syncthreads();
    for (int o = 128; o > 0; o >>= 1) {
        if (threadIdx.x < o) {
            sh0[threadIdx.x] += sh0[threadIdx.x + o];
            sh1[threadIdx.x] += sh1[threadIdx.x + o];
        }
        __syncthreads();
    }
    if (threadIdx.x == 0) {
        float mean = sh0[0] / (float)n;
        float var  = sh1[0] / (float)n - mean * mean;
        stats[2 * bg]     = mean;
        stats[2 * bg + 1] = rsqrtf(var + 1e-5f);
    }
}

// ---------------- K2: normalize + transpose to bf16 [B,N,C] ----------------
__global__ void norm_transpose_kernel(const float* __restrict__ x,
                                      const float* __restrict__ stats,
                                      const float* __restrict__ nw,
                                      const float* __restrict__ nb,
                                      bf16* __restrict__ hT) {
    const int b  = blockIdx.z;
    const int c0 = blockIdx.y * 32;
    const int n0 = blockIdx.x * 32;
    const int tx = threadIdx.x;   // 0..31 (fast: n on read, c on write)
    const int ty = threadIdx.y;   // 0..7
    __shared__ __align__(16) bf16 tile[32][33];
    #pragma unroll
    for (int it = 0; it < 4; ++it) {
        int c = c0 + ty + it * 8;
        float mean = stats[2 * (b * NGRP + (c >> 3))];
        float rstd = stats[2 * (b * NGRP + (c >> 3)) + 1];
        float v = x[(size_t)b * CH * NPOS + (size_t)c * NPOS + n0 + tx];
        tile[ty + it * 8][tx] = f2bf((v - mean) * rstd * nw[c] + nb[c]);
    }
    __syncthreads();
    #pragma unroll
    for (int it = 0; it < 4; ++it) {
        int n = n0 + ty + it * 8;
        hT[(size_t)b * NPOS * CH + (size_t)n * CH + c0 + tx] = tile[tx][ty + it * 8];
    }
}

// ---------------- K3: weights fp32 -> bf16 ----------------
__global__ void cvt_weights_kernel(const float* __restrict__ qkvw,
                                   const float* __restrict__ projw,
                                   bf16* __restrict__ wq,
                                   bf16* __restrict__ wp) {
    int i = blockIdx.x * 256 + threadIdx.x;
    if (i < 3 * CH * CH) wq[i] = f2bf(qkvw[i]);
    int j = i - 3 * CH * CH;
    if (j >= 0 && j < CH * CH) wp[j] = f2bf(projw[j]);
}

// ---------------- K4: QKV GEMM (WMMA) ----------------
// qkv_t[n,d] = sum_c hT[n,c] * wqkv[d,c] + b[d]
// store q,k as [B,N,C] bf16; v as [B,C,N] bf16 (so P*V B-frags are contiguous)
__global__ void qkv_gemm_kernel(const bf16* __restrict__ hT,
                                const bf16* __restrict__ wq,
                                const float* __restrict__ qkvb,
                                bf16* __restrict__ qT,
                                bf16* __restrict__ kT,
                                bf16* __restrict__ vM) {
    const int b    = blockIdx.y;
    const int wave = blockIdx.x * 4 + (threadIdx.x >> 5);
    const int lane = threadIdx.x & 31;
    const int NT   = NPOS / 16;                 // 256 n-tiles
    const int ti   = wave % NT;                 // n tile
    const int td   = wave / NT;                 // d tile (0..47)
    const bf16* A  = hT + (size_t)b * NPOS * CH + (size_t)(ti * 16) * CH;
    const bf16* Bp = wq + (size_t)(td * 16) * CH;
    v8f acc = vzero8();
    #pragma unroll
    for (int kc = 0; kc < CH; kc += 32)
        acc = wmma_bf16(load_frag_a(A + kc, CH), load_frag_b(Bp + kc, CH), acc);
    const int col = lane & 15, half = lane >> 4;
    const int d   = td * 16 + col;
    const float bias = qkvb[d];
    #pragma unroll
    for (int r = 0; r < 8; ++r) {
        int n = ti * 16 + half * 8 + r;         // D layout: VGPR r -> row r + half*8
        bf16 hv = f2bf(acc[r] + bias);
        if (d < CH)
            qT[(size_t)b * NPOS * CH + (size_t)n * CH + d] = hv;
        else if (d < 2 * CH)
            kT[(size_t)b * NPOS * CH + (size_t)n * CH + (d - CH)] = hv;
        else
            vM[(size_t)b * CH * NPOS + (size_t)(d - 2 * CH) * NPOS + n] = hv;
    }
}

// ---------------- K5: flash attention (WMMA, online softmax) ----------------
// one wave handles 16 query rows; streams K/V in 32-column chunks.
__global__ void flash_attn_kernel(const bf16* __restrict__ qT,
                                  const bf16* __restrict__ kT,
                                  const bf16* __restrict__ vM,
                                  bf16* __restrict__ aT) {
    const int wave = blockIdx.x * 4 + (threadIdx.x >> 5);
    const int lane = threadIdx.x & 31;
    const int b    = wave >> 8;                 // NPOS/16 = 256 waves per batch
    const int i0   = (wave & 255) * 16;
    const int col  = lane & 15, half = lane >> 4;

    // preload the 16 q-rows as 8 A-fragments (K = 0..255 in steps of 32)
    const bf16* Qp = qT + (size_t)b * NPOS * CH + (size_t)i0 * CH;
    v16bf qf[8];
    #pragma unroll
    for (int s = 0; s < 8; ++s) qf[s] = load_frag_a(Qp + s * 32, CH);

    v8f acc[16];                                // 16 c-tiles x 8 rows
    #pragma unroll
    for (int t = 0; t < 16; ++t) acc[t] = vzero8();
    float mrow[8], lrow[8];
    #pragma unroll
    for (int r = 0; r < 8; ++r) { mrow[r] = -__builtin_inff(); lrow[r] = 0.f; }

    __shared__ __align__(32) bf16 pbuf[4][16][32];
    bf16 (*pb)[32] = pbuf[threadIdx.x >> 5];

    for (int j0 = 0; j0 < NPOS; j0 += 32) {
        const bf16* Kp = kT + (size_t)b * NPOS * CH + (size_t)j0 * CH;
        v8f s0 = vzero8(), s1 = vzero8();
        #pragma unroll
        for (int s = 0; s < 8; ++s) {
            s0 = wmma_bf16(qf[s], load_frag_b(Kp + s * 32, CH), s0);
            s1 = wmma_bf16(qf[s], load_frag_b(Kp + (size_t)16 * CH + s * 32, CH), s1);
        }
        // scale + per-row max (rows live per 16-lane group in D layout)
        float rmax[8];
        #pragma unroll
        for (int r = 0; r < 8; ++r) {
            s0[r] *= SCALE; s1[r] *= SCALE;
            rmax[r] = fmaxf(s0[r], s1[r]);
        }
        for (int off = 1; off < 16; off <<= 1) {
            #pragma unroll
            for (int r = 0; r < 8; ++r)
                rmax[r] = fmaxf(rmax[r], __shfl_xor(rmax[r], off));
        }
        float p0[8], p1[8], rsum[8];
        #pragma unroll
        for (int r = 0; r < 8; ++r) {
            float mn   = fmaxf(mrow[r], rmax[r]);
            float corr = __expf(mrow[r] - mn);
            mrow[r] = mn;
            lrow[r] *= corr;
            #pragma unroll
            for (int t = 0; t < 16; ++t) acc[t][r] *= corr;
            p0[r]  = __expf(s0[r] - mn);
            p1[r]  = __expf(s1[r] - mn);
            rsum[r] = p0[r] + p1[r];
        }
        for (int off = 1; off < 16; off <<= 1) {
            #pragma unroll
            for (int r = 0; r < 8; ++r) rsum[r] += __shfl_xor(rsum[r], off);
        }
        // D-layout -> A-fragment layout for P via LDS round trip
        #pragma unroll
        for (int r = 0; r < 8; ++r) {
            lrow[r] += rsum[r];
            pb[half * 8 + r][col]      = f2bf(p0[r]);
            pb[half * 8 + r][16 + col] = f2bf(p1[r]);
        }
        v16bf pa = load_frag_a(&pb[0][0], 32);
        const bf16* Vp = vM + (size_t)b * CH * NPOS + j0;
        #pragma unroll
        for (int t = 0; t < 16; ++t)
            acc[t] = wmma_bf16(pa, load_frag_b(Vp + (size_t)(t * 16) * NPOS, NPOS), acc[t]);
    }
    // normalize and store a^T[B,N,C] in bf16 (feeds proj B-fragments)
    #pragma unroll
    for (int r = 0; r < 8; ++r) {
        float inv = 1.0f / lrow[r];
        int n = i0 + half * 8 + r;
        #pragma unroll
        for (int t = 0; t < 16; ++t)
            aT[(size_t)b * NPOS * CH + (size_t)n * CH + t * 16 + col] =
                f2bf(acc[t][r] * inv);
    }
}

// ---------------- K6: proj GEMM + bias + residual (WMMA) ----------------
// out[b,d,n] = x[b,d,n] + sum_c projw[d,c] * aT[b,n,c] + projb[d]
__global__ void proj_gemm_kernel(const bf16* __restrict__ aT,
                                 const bf16* __restrict__ wp,
                                 const float* __restrict__ projb,
                                 const float* __restrict__ x,
                                 float* __restrict__ out) {
    const int b    = blockIdx.y;
    const int wave = blockIdx.x * 4 + (threadIdx.x >> 5);
    const int lane = threadIdx.x & 31;
    const int NT   = NPOS / 16;
    const int ti   = wave % NT;                 // n tile
    const int td   = wave / NT;                 // d tile (0..15)
    const bf16* A  = wp + (size_t)(td * 16) * CH;                      // [d,c]
    const bf16* Bp = aT + (size_t)b * NPOS * CH + (size_t)(ti * 16) * CH; // [n,c]
    v8f acc = vzero8();
    #pragma unroll
    for (int kc = 0; kc < CH; kc += 32)
        acc = wmma_bf16(load_frag_a(A + kc, CH), load_frag_b(Bp + kc, CH), acc);
    const int col = lane & 15, half = lane >> 4;
    const int n = ti * 16 + col;
    #pragma unroll
    for (int r = 0; r < 8; ++r) {
        int d = td * 16 + half * 8 + r;
        size_t idx = (size_t)b * CH * NPOS + (size_t)d * NPOS + n;
        out[idx] = x[idx] + acc[r] + projb[d];
    }
}

// ---------------- launch ----------------
extern "C" void kernel_launch(void* const* d_in, const int* in_sizes, int n_in,
                              void* d_out, int out_size, void* d_ws, size_t ws_size,
                              hipStream_t stream) {
    const float* x      = (const float*)d_in[0];
    const float* norm_w = (const float*)d_in[1];
    const float* norm_b = (const float*)d_in[2];
    const float* qkv_w  = (const float*)d_in[3];
    const float* qkv_b  = (const float*)d_in[4];
    const float* proj_w = (const float*)d_in[5];
    const float* proj_b = (const float*)d_in[6];
    float* out = (float*)d_out;

    // workspace layout (bytes); total ~84.5 MB
    char* ws = (char*)d_ws;
    const size_t SZ_BNC = (size_t)BATCH * NPOS * CH * sizeof(bf16);  // 16 MiB
    float* stats = (float*)ws;                               // 2 KiB
    bf16*  hT    = (bf16*)(ws + 4096);
    bf16*  qT    = (bf16*)(ws + 4096 + 1 * SZ_BNC);
    bf16*  kT    = (bf16*)(ws + 4096 + 2 * SZ_BNC);
    bf16*  vM    = (bf16*)(ws + 4096 + 3 * SZ_BNC);
    bf16*  aT    = (bf16*)(ws + 4096 + 4 * SZ_BNC);
    bf16*  wqkv  = (bf16*)(ws + 4096 + 5 * SZ_BNC);
    bf16*  wproj = (bf16*)(ws + 4096 + 5 * SZ_BNC + (size_t)3 * CH * CH * sizeof(bf16));

    gn_stats_kernel<<<BATCH * NGRP, 256, 0, stream>>>(x, stats);
    norm_transpose_kernel<<<dim3(NPOS / 32, CH / 32, BATCH), dim3(32, 8), 0, stream>>>(
        x, stats, norm_w, norm_b, hT);
    cvt_weights_kernel<<<(4 * CH * CH) / 256, 256, 0, stream>>>(qkv_w, proj_w, wqkv, wproj);
    // 48 d-tiles * 256 n-tiles = 12288 waves/batch, 4 waves/block
    qkv_gemm_kernel<<<dim3(3072, BATCH), 128, 0, stream>>>(hT, wqkv, qkv_b, qT, kT, vM);
    // 256 waves/batch * 8 batches = 2048 waves, 4 waves/block
    flash_attn_kernel<<<512, 128, 0, stream>>>(qT, kT, vM, aT);
    // 16 d-tiles * 256 n-tiles = 4096 waves/batch
    proj_gemm_kernel<<<dim3(1024, BATCH), 128, 0, stream>>>(aT, wproj, proj_b, x, out);
}
